// JukeboxBottleneckBlock_87376814670611
// MI455X (gfx1250) — matrix operand order repaired and checked by reading, not compiled, not run on hardware
//
#include <hip/hip_runtime.h>
#include <hip/hip_bf16.h>
#include <math.h>

typedef __bf16 bf16;
typedef bf16  v16bf __attribute__((ext_vector_type(16)));
typedef float v8f   __attribute__((ext_vector_type(8)));

#define NB    8
#define WD    64
#define TT    8192
#define NN    (NB * TT)        // 65536 rows of x
#define KK    2048             // codebook entries
#define NELEM ((double)NN * (double)WD)

// ---------------------------------------------------------------------------
// Kernel 1: codebook -> bf16 hi/lo split + per-row squared norm
// ---------------------------------------------------------------------------
__global__ __launch_bounds__(64) void prep_codebook(const float* __restrict__ cb,
                                                    bf16* __restrict__ cb_hi,
                                                    bf16* __restrict__ cb_lo,
                                                    float* __restrict__ cnorm) {
    const int k = blockIdx.x;
    const int w = threadIdx.x;
    const float v = cb[k * WD + w];
    const bf16 h = (bf16)v;
    cb_hi[k * WD + w] = h;
    cb_lo[k * WD + w] = (bf16)(v - (float)h);
    float p = v * v;
    #pragma unroll
    for (int mask = 1; mask <= 16; mask <<= 1) p += __shfl_xor(p, mask, 32);
    __shared__ float s[2];
    if ((threadIdx.x & 31) == 0) s[threadIdx.x >> 5] = p;
    __syncthreads();
    if (threadIdx.x == 0) cnorm[k] = s[0] + s[1];
}

// ---------------------------------------------------------------------------
// Kernel 2: fused GEMM (bf16x3 WMMA) + argmin + dq gather + reductions.
// Codebook tiles are double-buffered in LDS via async global->LDS copies:
// each 4KB tile is fetched ONCE per block (vs once per wave), cutting L2
// traffic 8x; WMMAs on tile k overlap the async fetch of tile k+1.
// ---------------------------------------------------------------------------
__global__ __launch_bounds__(256) void vq_main(const float*  __restrict__ hidden,
                                               const float*  __restrict__ codebook,
                                               const bf16*   __restrict__ cb_hi,
                                               const bf16*   __restrict__ cb_lo,
                                               const float*  __restrict__ cnorm,
                                               double*       __restrict__ acc,
                                               float*        __restrict__ out) {
    __shared__ float s_cn[KK];
    __shared__ __align__(16) char s_buf[2][4096];  // [hi 2KB | lo 2KB] per tile
    __shared__ int   s_tok[128];
    __shared__ float s_red[3];   // [0]=fit sum, [1]=sum x, [2]=sum x^2

    const int tid  = threadIdx.x;
    const int lane = tid & 31;
    const int wave = tid >> 5;
    const int nblk = blockIdx.x * 128;
    const int n0   = nblk + wave * 16;
    const int m    = lane & 15;
    const bool hih = (lane >= 16);

    // async-stage one 4KB codebook tile (hi+lo) into LDS buffer bsel:
    // thread -> one 16-byte chunk, one global_load_async_to_lds_b128 each.
    const int      st_half = (tid >= 128);
    const size_t   st_gof  = (size_t)(st_half ? (tid - 128) : tid) * 16;
    const unsigned st_lof  = (unsigned)(st_half ? 2048 + (tid - 128) * 16 : tid * 16);
    auto stage = [&](int kt, int bsel) {
        const char* g = (st_half ? (const char*)cb_lo : (const char*)cb_hi)
                        + (size_t)kt * 2048 + st_gof;
        unsigned long long ga = (unsigned long long)g;
        unsigned ldsa = (unsigned)(uintptr_t)(&s_buf[bsel][0]) + st_lof;
        asm volatile("global_load_async_to_lds_b128 %0, %1, off"
                     :: "v"(ldsa), "v"(ga) : "memory");
    };

    stage(0, 0);                                  // prefetch tile 0
    for (int i = tid; i < KK; i += 256) s_cn[i] = cnorm[i];
    if (tid < 3) s_red[tid] = 0.f;

    // ---- build A fragments (hi/lo bf16) straight from [B,W,T] layout ----
    // A 16x32 bf16, k order chosen here (B matches): lanes 0-15 row M=lane hold K {0..15};
    // lanes 16-31 row M=lane-16 hold K {16..31}. Chunk1 = +32 on K(w).
    const int row = n0 + m;
    const int bb  = row >> 13;
    const int tt  = row & (TT - 1);
    const float* xp = hidden + (size_t)bb * (WD * TT) + tt;
    const int wbase = hih ? 16 : 0;

    float xv[32];
    #pragma unroll
    for (int g = 0; g < 4; ++g) {
        const int ws0 = wbase + (g >> 1) * 32 + (g & 1) * 8;
        #pragma unroll
        for (int j = 0; j < 8; ++j)
            xv[g * 8 + j] = xp[(size_t)(ws0 + j) * TT];
    }

    v16bf a0h, a0l, a1h, a1l;
    #pragma unroll
    for (int i = 0; i < 16; ++i) {
        const float v0 = xv[i], v1 = xv[16 + i];
        const bf16 h0 = (bf16)v0, h1 = (bf16)v1;
        a0h[i] = h0;
        a1h[i] = h1;
        a0l[i] = (bf16)(v0 - (float)h0);
        a1l[i] = (bf16)(v1 - (float)h1);
    }
    float xsum = 0.f, xsq = 0.f;
    #pragma unroll
    for (int i = 0; i < 32; ++i) { xsum += xv[i]; xsq += xv[i] * xv[i]; }

    float minv[8];
    int   mink[8];
    #pragma unroll
    for (int r = 0; r < 8; ++r) { minv[r] = 3.4e38f; mink[r] = 0; }

    asm volatile("s_wait_asynccnt 0x0" ::: "memory");
    __syncthreads();                              // tile 0 + s_cn visible

    // ---- main loop: 128 codebook tiles of 16 codes, double-buffered ----
    const int boff = hih ? 16 : 0;   // B 32x16 bf16, same k order as A: lanes>=16 hold K=16..31
    for (int kt = 0; kt < 128; ++kt) {
        const int cur = kt & 1;
        if (kt + 1 < 128) stage(kt + 1, cur ^ 1); // overlap fetch with WMMA

        const bf16* lh = (const bf16*)(&s_buf[cur][0]);
        const bf16* ll = lh + 1024;
        const v16bf b0h = *(const v16bf*)(lh + m * WD + boff);
        const v16bf b1h = *(const v16bf*)(lh + m * WD + boff + 32);
        const v16bf b0l = *(const v16bf*)(ll + m * WD + boff);
        const v16bf b1l = *(const v16bf*)(ll + m * WD + boff + 32);

        v8f acc0 = {};
        v8f acc1 = {};
        // two independent 3-deep chains (hi*hi + hi*lo + lo*hi per K-chunk)
        acc0 = __builtin_amdgcn_wmma_f32_16x16x32_bf16(false, a0h, false, b0h, (short)0, acc0, false, false);
        acc1 = __builtin_amdgcn_wmma_f32_16x16x32_bf16(false, a1h, false, b1h, (short)0, acc1, false, false);
        acc0 = __builtin_amdgcn_wmma_f32_16x16x32_bf16(false, a0h, false, b0l, (short)0, acc0, false, false);
        acc1 = __builtin_amdgcn_wmma_f32_16x16x32_bf16(false, a1h, false, b1l, (short)0, acc1, false, false);
        acc0 = __builtin_amdgcn_wmma_f32_16x16x32_bf16(false, a0l, false, b0h, (short)0, acc0, false, false);
        acc1 = __builtin_amdgcn_wmma_f32_16x16x32_bf16(false, a1l, false, b1h, (short)0, acc1, false, false);

        const int   k0 = kt * 16;
        const float cn = s_cn[k0 + m];   // C/D column N == lane%16 -> code k0+m
        #pragma unroll
        for (int r = 0; r < 8; ++r) {
            const float d = cn - 2.0f * (acc0[r] + acc1[r]);
            if (d < minv[r]) { minv[r] = d; mink[r] = k0 + m; }
        }

        // my async writes for the next tile are done + everyone finished
        // reading buffer `cur` before it gets overwritten two tiles later
        asm volatile("s_wait_asynccnt 0x0" ::: "memory");
        __syncthreads();
    }

    // ---- argmin across the 16 columns (xor-shuffle inside each 16-lane half,
    //      lowest-k tie-break to match jnp.argmin) ----
    #pragma unroll
    for (int mask = 1; mask <= 8; mask <<= 1) {
        #pragma unroll
        for (int r = 0; r < 8; ++r) {
            const float ov = __shfl_xor(minv[r], mask, 32);
            const int   ok = __shfl_xor(mink[r], mask, 32);
            if (ov < minv[r] || (ov == minv[r] && ok < mink[r])) {
                minv[r] = ov; mink[r] = ok;
            }
        }
    }

    // ||x_row||^2 : this lane holds half the w's, partner lane^16 the rest
    const float xn = xsq + __shfl_xor(xsq, 16, 32);

    // one writer lane per row: lanes 0..7 -> rows 0..7 (r=lane),
    // lanes 24..31 -> rows 8..15 (r=lane-24); both have matching xn.
    int r_w = -1, rowM = 0;
    if (lane < 8)        { r_w = lane;      rowM = lane;      }
    else if (lane >= 24) { r_w = lane - 24; rowM = lane - 16; }

    float selv = 0.f; int selk = 0;
    #pragma unroll
    for (int r = 0; r < 8; ++r)
        if (r == r_w) { selv = minv[r]; selk = mink[r]; }

    float fitc = 0.f;
    if (r_w >= 0) {
        s_tok[wave * 16 + rowM] = selk;
        fitc = xn + selv;                 // min squared distance of this row
    }
    #pragma unroll
    for (int mask = 1; mask <= 16; mask <<= 1) {
        fitc += __shfl_xor(fitc, mask, 32);
        xsum += __shfl_xor(xsum, mask, 32);
        xsq  += __shfl_xor(xsq,  mask, 32);
    }
    if (lane == 0) {
        atomicAdd(&s_red[0], fitc);
        atomicAdd(&s_red[1], xsum);
        atomicAdd(&s_red[2], xsq);
    }
    __syncthreads();
    if (tid == 0) {
        atomicAdd(&acc[0], (double)s_red[1]);   // sum x
        atomicAdd(&acc[1], (double)s_red[2]);   // sum x^2
        atomicAdd(&acc[2], (double)s_red[0]);   // sum min-dist
    }

    // ---- outputs: tokens (as float) + dq = codebook[token] transposed back ----
    const int blk_b = nblk >> 13;
    const int blk_t = nblk & (TT - 1);
    if (tid < 128) out[nblk + tid] = (float)s_tok[tid];

    float* dq = out + NN;                 // dq_out is [B, W, T] right after tokens
    #pragma unroll 4
    for (int it = 0; it < 32; ++it) {
        const int idx = it * 256 + tid;   // 0..8191 over (w, n_local)
        const int w   = idx >> 7;
        const int nl  = idx & 127;
        dq[((size_t)(blk_b * WD + w) << 13) + blk_t + nl] =
            codebook[(size_t)s_tok[nl] * WD + w];
    }
}

// ---------------------------------------------------------------------------
// Kernel 3: finalize the three scalars
// ---------------------------------------------------------------------------
__global__ void vq_finalize(const double* __restrict__ acc,
                            float* __restrict__ scal) {
    const double n    = NELEM;
    const double mean = acc[0] / n;
    double var        = acc[1] / n - mean * mean;
    if (var < 0.0) var = 0.0;
    scal[0] = (float)(acc[2] / n);              // commit_loss = sum_min / (N*W)
    scal[1] = (float)(acc[2] / (double)NN);     // fit         = sum_min / N
    scal[2] = (float)sqrt(var);                 // prenorm
}

// ---------------------------------------------------------------------------
extern "C" void kernel_launch(void* const* d_in, const int* in_sizes, int n_in,
                              void* d_out, int out_size, void* d_ws, size_t ws_size,
                              hipStream_t stream) {
    const float* hidden   = (const float*)d_in[0];   // [8, 64, 8192] f32
    const float* codebook = (const float*)d_in[1];   // [2048, 64]  f32
    float* out = (float*)d_out;                      // tokens | dq | 3 scalars

    char* ws = (char*)d_ws;
    bf16*  cb_hi = (bf16*)(ws);                                 // 256 KB
    bf16*  cb_lo = (bf16*)(ws + (size_t)KK * WD * 2);           // 256 KB
    float* cnorm = (float*)(ws + (size_t)KK * WD * 4);          // 8 KB
    double* acc  = (double*)(ws + (size_t)KK * WD * 4 + KK * 4);// 24 B, 8-aligned

    hipMemsetAsync(acc, 0, 3 * sizeof(double), stream);
    prep_codebook<<<KK, 64, 0, stream>>>(codebook, cb_hi, cb_lo, cnorm);
    vq_main<<<NN / 128, 256, 0, stream>>>(hidden, codebook, cb_hi, cb_lo,
                                          cnorm, acc, out);
    vq_finalize<<<1, 1, 0, stream>>>(acc, out + NN + (size_t)NB * WD * TT);
}
